// BitLinear_23321672417338
// MI455X (gfx1250) — 8-bit matrix operand order repaired and checked by reading, compiled, not run on hardware
//
#include <hip/hip_runtime.h>
#include <hip/hip_bf16.h>
#include <cstdint>

typedef int v8i __attribute__((ext_vector_type(8)));

#define D_IN   4096
#define D_OUT  4096
#define ROWS   16384
#define BK     64
#define LDSS   80

__device__ __forceinline__ int pack4_q(float4 v, float s) {
    int a = (int)rintf(v.x * s); a = a < -128 ? -128 : (a > 127 ? 127 : a);
    int b = (int)rintf(v.y * s); b = b < -128 ? -128 : (b > 127 ? 127 : b);
    int c = (int)rintf(v.z * s); c = c < -128 ? -128 : (c > 127 ? 127 : c);
    int d = (int)rintf(v.w * s); d = d < -128 ? -128 : (d > 127 ? 127 : d);
    return (a & 0xFF) | ((b & 0xFF) << 8) | ((c & 0xFF) << 16) | ((d & 0xFF) << 24);
}

__global__ __launch_bounds__(256) void rmsnorm_quant_kernel(
    const float* __restrict__ x, const float* __restrict__ rms_w,
    int8_t* __restrict__ xq, float* __restrict__ inv_sx)
{
    __shared__ float red[256];
    const int row = blockIdx.x;
    const int tid = threadIdx.x;

    const float4* xr = (const float4*)(x + (size_t)row * D_IN + tid * 16);
    const float4* wr = (const float4*)(rms_w + tid * 16);

    float4 v0 = xr[0], v1 = xr[1], v2 = xr[2], v3 = xr[3];
    float4 g0 = wr[0], g1 = wr[1], g2 = wr[2], g3 = wr[3];

    float ss = v0.x*v0.x + v0.y*v0.y + v0.z*v0.z + v0.w*v0.w
             + v1.x*v1.x + v1.y*v1.y + v1.z*v1.z + v1.w*v1.w
             + v2.x*v2.x + v2.y*v2.y + v2.z*v2.z + v2.w*v2.w
             + v3.x*v3.x + v3.y*v3.y + v3.z*v3.z + v3.w*v3.w;

    red[tid] = ss;
    __syncthreads();
    for (int s = 128; s > 0; s >>= 1) {
        if (tid < s) red[tid] += red[tid + s];
        __syncthreads();
    }
    const float rstd = rsqrtf(red[0] * (1.0f / D_IN) + 1e-5f);
    __syncthreads();

    auto nrm = [rstd](float4 v, float4 g) {
        v.x *= rstd * g.x; v.y *= rstd * g.y;
        v.z *= rstd * g.z; v.w *= rstd * g.w;
        return v;
    };
    v0 = nrm(v0, g0); v1 = nrm(v1, g1); v2 = nrm(v2, g2); v3 = nrm(v3, g3);

    auto amax4 = [](float4 v) {
        return fmaxf(fmaxf(fabsf(v.x), fabsf(v.y)), fmaxf(fabsf(v.z), fabsf(v.w)));
    };
    red[tid] = fmaxf(fmaxf(amax4(v0), amax4(v1)), fmaxf(amax4(v2), amax4(v3)));
    __syncthreads();
    for (int s = 128; s > 0; s >>= 1) {
        if (tid < s) red[tid] = fmaxf(red[tid], red[tid + s]);
        __syncthreads();
    }
    const float max_abs = fmaxf(red[0], 1e-5f);
    const float scale   = 127.0f / max_abs;

    int4 packed = { pack4_q(v0, scale), pack4_q(v1, scale),
                    pack4_q(v2, scale), pack4_q(v3, scale) };
    ((int4*)(xq + (size_t)row * D_IN))[tid] = packed;

    if (tid == 0) inv_sx[row] = max_abs * (1.0f / 127.0f);
}

__global__ __launch_bounds__(256) void quant_w_kernel(
    const float* __restrict__ w, int8_t* __restrict__ wq)
{
    const size_t idx = (size_t)blockIdx.x * blockDim.x + threadIdx.x;
    float4 v = ((const float4*)w)[idx];
    ((int*)wq)[idx] = pack4_q(v, 1.0f);
}

__global__ __launch_bounds__(512, 1) void bitlinear_gemm_kernel(
    const int8_t* __restrict__ xq, const int8_t* __restrict__ wq,
    const float* __restrict__ inv_sx, const float* __restrict__ scale_w,
    float* __restrict__ out)
{
    __shared__ int8_t ldsB[2][256 * LDSS];

    const int tid    = threadIdx.x;
    const int lane   = tid & 31;
    const int wave   = tid >> 5;
    const int lane16 = lane & 15;
    const int half   = lane >> 4;

    const int waveM = wave & 3;
    const int waveN = wave >> 2;
    const int mBase = blockIdx.y * 256 + waveM * 64;
    const int nBlk  = blockIdx.x * 256;
    const int nBase = nBlk + waveN * 64;

    const int8_t* pA[4];
#pragma unroll
    for (int mt = 0; mt < 4; ++mt)
        pA[mt] = xq + (size_t)(mBase + mt * 16 + lane16) * D_IN + half * 16;

    const int fr = tid >> 1;
    const int fs = (tid & 1) * 32;
    const int8_t* pBg = wq + (size_t)(nBlk + fr) * D_IN + fs;
    int8_t* fill0 = &ldsB[0][fr * LDSS + fs];
    int8_t* fill1 = &ldsB[1][fr * LDSS + fs];

    const int8_t* rd0[4];
    const int8_t* rd1[4];
#pragma unroll
    for (int nt = 0; nt < 4; ++nt) {
        const int off = (waveN * 64 + nt * 16 + lane16) * LDSS + half * 16;
        rd0[nt] = &ldsB[0][off];
        rd1[nt] = &ldsB[1][off];
    }

    v8i acc[4][4] = {};

    {
        int4 q0 = *(const int4*)(pBg + 0);
        int4 q1 = *(const int4*)(pBg + 16);
        *(int4*)(fill0 + 0)  = q0;
        *(int4*)(fill0 + 16) = q1;
    }
    __syncthreads();

    const int NSTEP = D_IN / BK;
    for (int s = 0; s < NSTEP; ++s) {
        const int k0  = s * BK;
        const int cur = s & 1;
        const bool hasNext = (s + 1) < NSTEP;

        int4 n0, n1;
        if (hasNext) {
            n0 = *(const int4*)(pBg + k0 + BK);
            n1 = *(const int4*)(pBg + k0 + BK + 16);
        }

        v8i b[4];
        const int8_t* const* rd = cur ? rd1 : rd0;
#pragma unroll
        for (int nt = 0; nt < 4; ++nt) {
            int4 q0 = *(const int4*)(rd[nt]);
            int4 q1 = *(const int4*)(rd[nt] + 32);
            b[nt][0] = q0.x; b[nt][1] = q0.y; b[nt][2] = q0.z; b[nt][3] = q0.w;
            b[nt][4] = q1.x; b[nt][5] = q1.y; b[nt][6] = q1.z; b[nt][7] = q1.w;
        }

        v8i a[4];
#pragma unroll
        for (int mt = 0; mt < 4; ++mt) {
            const int8_t* p = pA[mt] + k0;
            int2 q0 = *(const int2*)(p);
            int2 q1 = *(const int2*)(p + 8);
            int2 q2 = *(const int2*)(p + 32);
            int2 q3 = *(const int2*)(p + 40);
            a[mt][0] = q0.x; a[mt][1] = q0.y;
            a[mt][2] = q1.x; a[mt][3] = q1.y;
            a[mt][4] = q2.x; a[mt][5] = q2.y;
            a[mt][6] = q3.x; a[mt][7] = q3.y;
            if (hasNext) __builtin_prefetch(p + BK, 0, 1);
        }

#pragma unroll
        for (int mt = 0; mt < 4; ++mt)
#pragma unroll
            for (int nt = 0; nt < 4; ++nt)
                acc[mt][nt] = __builtin_amdgcn_wmma_i32_16x16x64_iu8(
                     true, a[mt],  true, b[nt],
                    acc[mt][nt],  false,  false);

        __syncthreads();
        if (hasNext) {
            int8_t* dst = cur ? fill0 : fill1;
            *(int4*)(dst + 0)  = n0;
            *(int4*)(dst + 16) = n1;
            __syncthreads();
        }
    }

    const float inv_sw = 1.0f / scale_w[0];
#pragma unroll
    for (int mt = 0; mt < 4; ++mt) {
        const int rowBase = mBase + mt * 16 + half * 8;
#pragma unroll
        for (int r = 0; r < 8; ++r) {
            const int   row = rowBase + r;
            const float sc  = inv_sx[row] * inv_sw;
#pragma unroll
            for (int nt = 0; nt < 4; ++nt) {
                out[(size_t)row * D_OUT + nBase + nt * 16 + lane16] =
                    (float)acc[mt][nt][r] * sc;
            }
        }
    }
}

extern "C" void kernel_launch(void* const* d_in, const int* in_sizes, int n_in,
                              void* d_out, int out_size, void* d_ws, size_t ws_size,
                              hipStream_t stream) {
    const float* x     = (const float*)d_in[0];
    const float* rms_w = (const float*)d_in[1];
    const float* w     = (const float*)d_in[2];
    const float* sw    = (const float*)d_in[3];
    float* out = (float*)d_out;

    int8_t* xq     = (int8_t*)d_ws;
    int8_t* wq     = xq + (size_t)ROWS * D_IN;
    float*  inv_sx = (float*)(wq + (size_t)D_OUT * D_IN);

    rmsnorm_quant_kernel<<<ROWS, 256, 0, stream>>>(x, rms_w, xq, inv_sx);

    quant_w_kernel<<<(D_OUT * (size_t)D_IN / 4) / 256, 256, 0, stream>>>(w, wq);

    dim3 grid(D_OUT / 256, ROWS / 256);
    bitlinear_gemm_kernel<<<grid, 512, 0, stream>>>(xq, wq, inv_sx, sw, out);
}
